// Attention_14766097564112
// MI455X (gfx1250) — compile-verified, matrix operand order checked by reading, not run on hardware
//
#include <hip/hip_runtime.h>
#include <hip/hip_bf16.h>
#include <math.h>

// ---------------------------------------------------------------------------
// Types for CDNA5 WMMA (wave32)
// ---------------------------------------------------------------------------
typedef __attribute__((ext_vector_type(16))) __bf16   v16bf;
typedef __attribute__((ext_vector_type(8)))  float    v8f;
typedef __attribute__((ext_vector_type(8)))  unsigned short u16x8;

#define DIMC   768
#define HEADS  12
#define HD     64
#define BATCH  8
#define SEQ    1024   // 32*32
#define MROWS  (BATCH*SEQ)   // 8192
#define QKVN   (3*DIMC)      // 2304
#define FFN    (4*DIMC)      // 3072

static __device__ __forceinline__ unsigned short f2bf(float f) {
    unsigned int u = __float_as_uint(f);
    u += 0x7fffu + ((u >> 16) & 1u);   // round-to-nearest-even
    return (unsigned short)(u >> 16);
}

// Load one 16x32 bf16 WMMA A/B fragment from LDS (row-major, stride in ushorts).
// Lane L (<16): elems 0..7 -> K=kbase+0..7,  elems 8..15 -> K=kbase+16..23  (row M/N = row0+L)
// Lane L (>=16): +8 on both K ranges.
static __device__ __forceinline__ v16bf load_frag(const unsigned short* base, int stride,
                                                  int row0, int kbase, int lane) {
    int r  = row0 + (lane & 15);
    int hf = lane >> 4;
    const unsigned short* p = base + r * stride + kbase + hf * 8;
    union { v16bf v; u16x8 u[2]; } f;
    f.u[0] = *(const u16x8*)(p);
    f.u[1] = *(const u16x8*)(p + 16);
    return f.v;
}

// ---------------------------------------------------------------------------
// GEMM: C[M,N] = A[M,K](bf16) * B[K,N]  (B stored transposed: Bt[N,K] bf16)
// Block tile 128x128, 8 waves (4x2), wave tile 32x64, K-step 64
// (16 WMMAs per barrier epoch).
// ---------------------------------------------------------------------------
enum { EPI_F32 = 0, EPI_BF16 = 1, EPI_F32_RES = 2, EPI_BF16_GELU = 3 };

template<int EPI>
__global__ __launch_bounds__(256) void gemm_bf16_k(
    const unsigned short* __restrict__ A,
    const unsigned short* __restrict__ Bt,
    const float* __restrict__ bias,
    const float* __restrict__ alpha,   // residual scale (EPI_F32_RES)
    const float* __restrict__ res,     // residual [M,N] f32 (EPI_F32_RES)
    float* __restrict__ Cf,
    unsigned short* __restrict__ Cb,
    int Mi, int Ni, int Ki)
{
    __shared__ __align__(16) unsigned short As[128 * 72];   // 128 rows x 64 K (+8 pad)
    __shared__ __align__(16) unsigned short Bs[128 * 72];

    const int tid  = threadIdx.x;
    const int lane = tid & 31;
    const int wave = tid >> 5;
    const int wm   = wave >> 1;          // 0..3
    const int wn   = wave & 1;           // 0..1
    const int hf   = lane >> 4;
    const int mBlk = blockIdx.y * 128;
    const int nBlk = blockIdx.x * 128;

    v8f acc[2][4] = {};

    for (int k0 = 0; k0 < Ki; k0 += 64) {
        __syncthreads();
#pragma unroll
        for (int i = 0; i < 4; ++i) {
            int cid = i * 256 + tid;
            int row = cid >> 3, ch = cid & 7;   // 128 rows x 8 chunks of 8 ushorts
            const unsigned short* ga = A  + (size_t)(mBlk + row) * Ki + k0 + ch * 8;
            const unsigned short* gb = Bt + (size_t)(nBlk + row) * Ki + k0 + ch * 8;
            *(uint4*)(&As[row * 72 + ch * 8]) = *(const uint4*)ga;
            *(uint4*)(&Bs[row * 72 + ch * 8]) = *(const uint4*)gb;
            if (k0 + 64 < Ki) {                 // global_prefetch_b8 of next K tile
                __builtin_prefetch(ga + 64, 0, 1);
                __builtin_prefetch(gb + 64, 0, 1);
            }
        }
        __syncthreads();

#pragma unroll
        for (int kk = 0; kk < 2; ++kk) {        // two 32-wide K chunks
            v16bf af[2], bfm[4];
            af[0] = load_frag(As, 72, wm * 32,      kk * 32, lane);
            af[1] = load_frag(As, 72, wm * 32 + 16, kk * 32, lane);
#pragma unroll
            for (int j = 0; j < 4; ++j)
                bfm[j] = load_frag(Bs, 72, wn * 64 + j * 16, kk * 32, lane);
#pragma unroll
            for (int i = 0; i < 2; ++i)
#pragma unroll
                for (int j = 0; j < 4; ++j)
                    acc[i][j] = __builtin_amdgcn_wmma_f32_16x16x32_bf16(
                        false, af[i], false, bfm[j], (short)0, acc[i][j], false, false);
        }
    }

    // Epilogue. C-layout: lane holds N = lane&15, row = r + 8*(lane>>4).
#pragma unroll
    for (int i = 0; i < 2; ++i)
#pragma unroll
        for (int j = 0; j < 4; ++j)
#pragma unroll
            for (int r = 0; r < 8; ++r) {
                int m = mBlk + wm * 32 + i * 16 + r + 8 * hf;
                int n = nBlk + wn * 64 + j * 16 + (lane & 15);
                float v = acc[i][j][r] + bias[n];
                size_t o = (size_t)m * Ni + n;
                if constexpr (EPI == EPI_F32_RES) {
                    Cf[o] = v + alpha[n] * res[o];
                } else if constexpr (EPI == EPI_BF16) {
                    Cb[o] = f2bf(v);
                } else if constexpr (EPI == EPI_BF16_GELU) {
                    float g = 0.5f * v * (1.0f + erff(v * 0.70710678118654752f));
                    Cb[o] = f2bf(g);
                } else {
                    Cf[o] = v;
                }
            }
}

// ---------------------------------------------------------------------------
// Streaming sigmoid attention: O = sum_kt sigmoid(Q Kt^T / 8) * Vt
// grid: (qblocks=8, heads=12, batch=8); 256 threads; 128 q-rows per block,
// 16 key tiles of 64. Each wave owns 16 q rows; Z bounced via private LDS.
// Sigmoid uses native v_exp_f32 / v_rcp_f32.
// ---------------------------------------------------------------------------
__global__ __launch_bounds__(256) void attn_sigmoid_k(
    const unsigned short* __restrict__ QKV,   // [MROWS, 2304] bf16
    unsigned short* __restrict__ Obf,         // [MROWS, 768]  bf16
    float inv_scale)
{
    __shared__ __align__(16) unsigned short Qs[128 * 72];
    __shared__ __align__(16) unsigned short Ks[64 * 72];
    __shared__ __align__(16) unsigned short Vt[64 * 72];     // Vt[d][key]
    __shared__ __align__(16) unsigned short Zs[8 * 16 * 72]; // per-wave 16x64

    const int tid  = threadIdx.x;
    const int lane = tid & 31;
    const int wave = tid >> 5;
    const int hf   = lane >> 4;
    const int qb = blockIdx.x, h = blockIdx.y, b = blockIdx.z;
    const size_t qrow0 = (size_t)b * SEQ + qb * 128;
    const int qoff = h * HD, koff = DIMC + h * HD, voff = 2 * DIMC + h * HD;

    // Q tile: 128 rows x 64 bf16
#pragma unroll
    for (int i = 0; i < 4; ++i) {
        int cid = i * 256 + tid;
        int row = cid >> 3, ch = cid & 7;
        *(uint4*)(&Qs[row * 72 + ch * 8]) =
            *(const uint4*)(QKV + (qrow0 + row) * QKVN + qoff + ch * 8);
    }

    v8f oacc[4] = {};
    unsigned short* Zw = &Zs[wave * 16 * 72];

    for (int kt = 0; kt < 16; ++kt) {
        __syncthreads();
        size_t krow0 = (size_t)b * SEQ + kt * 64;
#pragma unroll
        for (int i = 0; i < 2; ++i) {
            int cid = i * 256 + tid;
            int row = cid >> 3, ch = cid & 7;
            *(uint4*)(&Ks[row * 72 + ch * 8]) =
                *(const uint4*)(QKV + (krow0 + row) * QKVN + koff + ch * 8);
        }
#pragma unroll
        for (int i = 0; i < 16; ++i) {       // V transpose into LDS
            int idx = i * 256 + tid;
            int key = idx >> 6, d = idx & 63;
            Vt[d * 72 + key] = QKV[(krow0 + key) * QKVN + voff + d];
        }
        __syncthreads();

        v16bf aq0 = load_frag(Qs, 72, wave * 16, 0,  lane);
        v16bf aq1 = load_frag(Qs, 72, wave * 16, 32, lane);

        // S = Q * K^T (K tile rows are already Bt[N=key][K=d] layout)
#pragma unroll
        for (int sb = 0; sb < 4; ++sb) {
            v8f s = {};
            v16bf bk0 = load_frag(Ks, 72, sb * 16, 0,  lane);
            v16bf bk1 = load_frag(Ks, 72, sb * 16, 32, lane);
            s = __builtin_amdgcn_wmma_f32_16x16x32_bf16(false, aq0, false, bk0, (short)0, s, false, false);
            s = __builtin_amdgcn_wmma_f32_16x16x32_bf16(false, aq1, false, bk1, (short)0, s, false, false);
#pragma unroll
            for (int r = 0; r < 8; ++r) {
                // sigmoid via native exp/rcp: v_exp_f32 + v_rcp_f32
                float e  = __expf(-s[r] * inv_scale);
                float zv = __builtin_amdgcn_rcpf(1.0f + e);
                Zw[(r + 8 * hf) * 72 + sb * 16 + (lane & 15)] = f2bf(zv);
            }
        }
        // O += Z * V   (A = Z rows, Bt = Vt[d][key])
#pragma unroll
        for (int kk = 0; kk < 2; ++kk) {
            v16bf za = load_frag(Zw, 72, 0, kk * 32, lane);
#pragma unroll
            for (int j = 0; j < 4; ++j) {
                v16bf vb = load_frag(Vt, 72, j * 16, kk * 32, lane);
                oacc[j] = __builtin_amdgcn_wmma_f32_16x16x32_bf16(
                    false, za, false, vb, (short)0, oacc[j], false, false);
            }
        }
    }

#pragma unroll
    for (int j = 0; j < 4; ++j)
#pragma unroll
        for (int r = 0; r < 8; ++r) {
            size_t m = qrow0 + wave * 16 + r + 8 * hf;
            int n = h * HD + j * 16 + (lane & 15);
            Obf[m * DIMC + n] = f2bf(oacc[j][r]);
        }
}

// ---------------------------------------------------------------------------
// LayerNorm over C=768 per row; optional f32 and bf16 outputs.
// ---------------------------------------------------------------------------
__global__ __launch_bounds__(256) void layernorm_k(
    const float* __restrict__ x, const float* __restrict__ g, const float* __restrict__ bia,
    float* __restrict__ outf, unsigned short* __restrict__ outb)
{
    __shared__ float r1[256];
    __shared__ float r2[256];
    const int row = blockIdx.x;
    const float* xr = x + (size_t)row * DIMC;
    float v[3], s = 0.f, s2 = 0.f;
#pragma unroll
    for (int i = 0; i < 3; ++i) {
        v[i] = xr[threadIdx.x + i * 256];
        s += v[i]; s2 += v[i] * v[i];
    }
    r1[threadIdx.x] = s; r2[threadIdx.x] = s2;
    __syncthreads();
    for (int off = 128; off > 0; off >>= 1) {
        if (threadIdx.x < off) {
            r1[threadIdx.x] += r1[threadIdx.x + off];
            r2[threadIdx.x] += r2[threadIdx.x + off];
        }
        __syncthreads();
    }
    const float mean = r1[0] * (1.0f / DIMC);
    const float var  = r2[0] * (1.0f / DIMC) - mean * mean;
    const float rstd = rsqrtf(var + 1e-5f);
#pragma unroll
    for (int i = 0; i < 3; ++i) {
        int c = threadIdx.x + i * 256;
        float y = (v[i] - mean) * rstd * g[c] + bia[c];
        size_t o = (size_t)row * DIMC + c;
        if (outf) outf[o] = y;
        if (outb) outb[o] = f2bf(y);
    }
}

// ---------------------------------------------------------------------------
// Depthwise conv KSxKS over NHWC-flat [B, H*W, C] activations.
// ---------------------------------------------------------------------------
template<int KS>
__global__ __launch_bounds__(256) void dwconv_k(
    const float* __restrict__ x, const float* __restrict__ w, const float* __restrict__ bia,
    float* __restrict__ out, int Bn, int Hh, int Wn, int Cc)
{
    long idx = (long)blockIdx.x * 256 + threadIdx.x;
    long total = (long)Bn * Hh * Wn * Cc;
    if (idx >= total) return;
    int c = idx % Cc; long t = idx / Cc;
    int ww = t % Wn; t /= Wn;
    int hh = t % Hh; int bb = t / Hh;
    const int P = KS / 2;
    float acc = bia[c];
#pragma unroll
    for (int dy = 0; dy < KS; ++dy) {
        int y = hh + dy - P;
        if (y < 0 || y >= Hh) continue;
#pragma unroll
        for (int dx = 0; dx < KS; ++dx) {
            int xw = ww + dx - P;
            if (xw < 0 || xw >= Wn) continue;
            acc += x[((long)bb * Hh * Wn + (long)y * Wn + xw) * Cc + c]
                 * w[c * KS * KS + dy * KS + dx];
        }
    }
    out[idx] = acc;
}

// ---------------------------------------------------------------------------
// Layout / packing utility kernels
// ---------------------------------------------------------------------------
__global__ __launch_bounds__(256) void nchw_to_nhwc_k(
    const float* __restrict__ in, float* __restrict__ outf, unsigned short* __restrict__ outb,
    int Bn, int Cc, int HW)
{
    long idx = (long)blockIdx.x * 256 + threadIdx.x;
    long total = (long)Bn * Cc * HW;
    if (idx >= total) return;
    int c = idx % Cc;
    int n = (idx / Cc) % HW;
    int b = idx / ((long)Cc * HW);
    float v = in[((long)b * Cc + c) * HW + n];
    outf[idx] = v;
    outb[idx] = f2bf(v);
}

// y/x4 are NHWC-flat [B,N,C]; out is NCHW [B,C,H,W].
__global__ __launch_bounds__(256) void final_combine_k(
    const float* __restrict__ y2, const float* __restrict__ x4,
    const float* __restrict__ scale, const float* __restrict__ alphas,
    float* __restrict__ out, int Bn, int Cc, int HW)
{
    long idx = (long)blockIdx.x * 256 + threadIdx.x;
    long total = (long)Bn * Cc * HW;
    if (idx >= total) return;
    int n = idx % HW;
    int c = (idx / HW) % Cc;
    int b = idx / ((long)HW * Cc);
    long src = ((long)b * HW + n) * Cc + c;
    out[idx] = scale[c] * y2[src] + alphas[c] * x4[src];
}

// Pack per-head q/k/v weights [H,C,HD] into one transposed bf16 matrix
// Wt[2304][768] (n = part*768 + h*64 + d, k = channel); concat biases too.
__global__ __launch_bounds__(256) void pack_qkv_k(
    const float* __restrict__ wq, const float* __restrict__ wk, const float* __restrict__ wv,
    const float* __restrict__ bq, const float* __restrict__ bk, const float* __restrict__ bv,
    unsigned short* __restrict__ Wt, float* __restrict__ bias)
{
    long idx = (long)blockIdx.x * 256 + threadIdx.x;
    if (idx >= (long)QKVN * DIMC) return;
    int k = idx % DIMC;
    int n = idx / DIMC;
    int part = n / DIMC, nn = n % DIMC;
    int h = nn >> 6, d = nn & 63;
    const float* w = (part == 0) ? wq : (part == 1) ? wk : wv;
    Wt[idx] = f2bf(w[((long)h * DIMC + k) * HD + d]);
    if (k == 0) {
        const float* bb = (part == 0) ? bq : (part == 1) ? bk : bv;
        bias[n] = bb[h * HD + d];
    }
}

// W[K,N] f32 -> Wt[N,K] bf16
__global__ __launch_bounds__(256) void transpose_w_k(
    const float* __restrict__ W, unsigned short* __restrict__ Wt, int K, int N)
{
    long idx = (long)blockIdx.x * 256 + threadIdx.x;
    if (idx >= (long)K * N) return;
    int  k = idx % K;
    long n = idx / K;
    Wt[idx] = f2bf(W[(long)k * N + n]);
}

// ---------------------------------------------------------------------------
// Host orchestration
// ---------------------------------------------------------------------------
extern "C" void kernel_launch(void* const* d_in, const int* in_sizes, int n_in,
                              void* d_out, int out_size, void* d_ws, size_t ws_size,
                              hipStream_t stream)
{
    (void)in_sizes; (void)n_in; (void)out_size; (void)ws_size;

    const float* decoder = (const float*)d_in[0];
    const float* wq1 = (const float*)d_in[1];  const float* bq1 = (const float*)d_in[2];
    const float* wk1 = (const float*)d_in[3];  const float* bk1 = (const float*)d_in[4];
    const float* wv1 = (const float*)d_in[5];  const float* bv1 = (const float*)d_in[6];
    const float* wo1 = (const float*)d_in[7];  const float* bo1 = (const float*)d_in[8];
    const float* wq2 = (const float*)d_in[9];  const float* bq2 = (const float*)d_in[10];
    const float* wk2 = (const float*)d_in[11]; const float* bk2 = (const float*)d_in[12];
    const float* wv2 = (const float*)d_in[13]; const float* bv2 = (const float*)d_in[14];
    const float* wo2 = (const float*)d_in[15]; const float* bo2 = (const float*)d_in[16];
    const float* alphas1 = (const float*)d_in[17];
    const float* alphas2 = (const float*)d_in[18];
    const float* peg_w = (const float*)d_in[19]; const float* peg_b = (const float*)d_in[20];
    const float* ln1_g = (const float*)d_in[21]; const float* ln1_b = (const float*)d_in[22];
    const float* ln2_g = (const float*)d_in[23]; const float* ln2_b = (const float*)d_in[24];
    const float* cn_dw_w = (const float*)d_in[25]; const float* cn_dw_b = (const float*)d_in[26];
    const float* cn_ln_g = (const float*)d_in[27]; const float* cn_ln_b = (const float*)d_in[28];
    const float* cn_w1 = (const float*)d_in[29]; const float* cn_b1 = (const float*)d_in[30];
    const float* cn_w2 = (const float*)d_in[31]; const float* cn_b2 = (const float*)d_in[32];
    const float* cn_scale  = (const float*)d_in[33];
    const float* cn_alphas = (const float*)d_in[34];

    char* ws = (char*)d_ws;
    size_t off = 0;
    auto alloc = [&](size_t bytes) -> void* {
        off = (off + 255) & ~(size_t)255;
        void* p = ws + off;
        off += bytes;
        return p;
    };

    const size_t M = MROWS;
    unsigned short* wqkv1T = (unsigned short*)alloc((size_t)QKVN * DIMC * 2);
    float*          bqkv1  = (float*)alloc(QKVN * 4);
    unsigned short* wqkv2T = (unsigned short*)alloc((size_t)QKVN * DIMC * 2);
    float*          bqkv2  = (float*)alloc(QKVN * 4);
    unsigned short* wo1T   = (unsigned short*)alloc((size_t)DIMC * DIMC * 2);
    unsigned short* wo2T   = (unsigned short*)alloc((size_t)DIMC * DIMC * 2);
    unsigned short* cnw1T  = (unsigned short*)alloc((size_t)FFN * DIMC * 2);
    unsigned short* cnw2T  = (unsigned short*)alloc((size_t)DIMC * FFN * 2);
    float*          x0f   = (float*)alloc(M * DIMC * 4);
    unsigned short* x0b   = (unsigned short*)alloc(M * DIMC * 2);
    unsigned short* qkvb  = (unsigned short*)alloc(M * QKVN * 2);
    unsigned short* obf   = (unsigned short*)alloc(M * DIMC * 2);
    float*          x1f   = (float*)alloc(M * DIMC * 4);
    float*          convf = (float*)alloc(M * DIMC * 4);
    float*          ln1f  = (float*)alloc(M * DIMC * 4);
    unsigned short* ln1b  = (unsigned short*)alloc(M * DIMC * 2);
    float*          x2f   = (float*)alloc(M * DIMC * 4);
    float*          ln2f  = (float*)alloc(M * DIMC * 4);
    unsigned short* lncnb = (unsigned short*)alloc(M * DIMC * 2);
    unsigned short* gelub = (unsigned short*)alloc(M * FFN * 2);
    float*          y2f   = (float*)alloc(M * DIMC * 4);

    const dim3 thr(256);
    const int elemBlocks = (int)((M * DIMC) / 256);          // 24576
    const float inv_scale = 0.125f;                          // 1/sqrt(64)

    // ---- weight prep ----
    pack_qkv_k<<<(QKVN * DIMC) / 256, thr, 0, stream>>>(wq1, wk1, wv1, bq1, bk1, bv1, wqkv1T, bqkv1);
    pack_qkv_k<<<(QKVN * DIMC) / 256, thr, 0, stream>>>(wq2, wk2, wv2, bq2, bk2, bv2, wqkv2T, bqkv2);
    transpose_w_k<<<(DIMC * DIMC) / 256, thr, 0, stream>>>(wo1, wo1T, DIMC, DIMC);
    transpose_w_k<<<(DIMC * DIMC) / 256, thr, 0, stream>>>(wo2, wo2T, DIMC, DIMC);
    transpose_w_k<<<(DIMC * FFN) / 256, thr, 0, stream>>>(cn_w1, cnw1T, DIMC, FFN);
    transpose_w_k<<<(DIMC * FFN) / 256, thr, 0, stream>>>(cn_w2, cnw2T, FFN, DIMC);

    // ---- x0 = NHWC(decoder) ----
    nchw_to_nhwc_k<<<elemBlocks, thr, 0, stream>>>(decoder, x0f, x0b, BATCH, DIMC, SEQ);

    // ---- block 1: QKV -> attention -> proj(+alpha1*x0) ----
    gemm_bf16_k<EPI_BF16><<<dim3(QKVN / 128, MROWS / 128), thr, 0, stream>>>(
        x0b, wqkv1T, bqkv1, nullptr, nullptr, nullptr, qkvb, MROWS, QKVN, DIMC);
    attn_sigmoid_k<<<dim3(SEQ / 128, HEADS, BATCH), thr, 0, stream>>>(qkvb, obf, inv_scale);
    gemm_bf16_k<EPI_F32_RES><<<dim3(DIMC / 128, MROWS / 128), thr, 0, stream>>>(
        obf, wo1T, bo1, alphas1, x0f, x1f, nullptr, MROWS, DIMC, DIMC);

    // ---- PEG 3x3 dwconv -> LN1 ----
    dwconv_k<3><<<elemBlocks, thr, 0, stream>>>(x1f, peg_w, peg_b, convf, BATCH, 32, 32, DIMC);
    layernorm_k<<<MROWS, thr, 0, stream>>>(convf, ln1_g, ln1_b, ln1f, ln1b);

    // ---- block 2 ----
    gemm_bf16_k<EPI_BF16><<<dim3(QKVN / 128, MROWS / 128), thr, 0, stream>>>(
        ln1b, wqkv2T, bqkv2, nullptr, nullptr, nullptr, qkvb, MROWS, QKVN, DIMC);
    attn_sigmoid_k<<<dim3(SEQ / 128, HEADS, BATCH), thr, 0, stream>>>(qkvb, obf, inv_scale);
    gemm_bf16_k<EPI_F32_RES><<<dim3(DIMC / 128, MROWS / 128), thr, 0, stream>>>(
        obf, wo2T, bo2, alphas2, ln1f, x2f, nullptr, MROWS, DIMC, DIMC);

    // ---- LN2 -> CNBlock head ----
    layernorm_k<<<MROWS, thr, 0, stream>>>(x2f, ln2_g, ln2_b, ln2f, nullptr);
    dwconv_k<7><<<elemBlocks, thr, 0, stream>>>(ln2f, cn_dw_w, cn_dw_b, convf, BATCH, 32, 32, DIMC);
    layernorm_k<<<MROWS, thr, 0, stream>>>(convf, cn_ln_g, cn_ln_b, nullptr, lncnb);
    gemm_bf16_k<EPI_BF16_GELU><<<dim3(FFN / 128, MROWS / 128), thr, 0, stream>>>(
        lncnb, cnw1T, cn_b1, nullptr, nullptr, nullptr, gelub, MROWS, FFN, DIMC);
    gemm_bf16_k<EPI_F32><<<dim3(DIMC / 128, MROWS / 128), thr, 0, stream>>>(
        gelub, cnw2T, cn_b2, nullptr, nullptr, y2f, nullptr, MROWS, DIMC, FFN);

    // ---- out[b,c,h,w] = cn_scale[c]*y + cn_alphas[c]*x4 ----
    final_combine_k<<<elemBlocks, thr, 0, stream>>>(
        y2f, ln2f, cn_scale, cn_alphas, (float*)d_out, BATCH, DIMC, SEQ);
}
